// FastGenomicBottleNet_80410377716332
// MI455X (gfx1250) — matrix operand order repaired and checked by reading, not compiled, not run on hardware
//
#include <hip/hip_runtime.h>
#include <stdint.h>

typedef __attribute__((ext_vector_type(16))) __bf16 v16bf;
typedef __attribute__((ext_vector_type(8)))  __bf16 v8bf;
typedef __attribute__((ext_vector_type(8)))  float  v8f;

#define B_DIM   128
#define T_TILES 16384
#define S0K     64              // K of stage-1 GEMM
#define S1N     64              // N columns per tile
#define LDS_ROW 72              // 64 bf16 + 8 pad -> 144 B row: 16B aligned, conflict-free
#define TPB     8               // tiles per block

// constant-mask lane XOR via ds_swizzle (group-of-32: and=0x1f, or=0, xor=MASK)
template <int MASK>
__device__ __forceinline__ float shflx(float v) {
  int i = __builtin_bit_cast(int, v);
  i = __builtin_amdgcn_ds_swizzle(i, (MASK << 10) | 0x1f);
  return __builtin_bit_cast(float, i);
}

__device__ __forceinline__ void load_tile(const float* __restrict__ W1, int t,
                                          int tid, float4 f[4]) {
  const float4* s4 = (const float4*)(W1 + (size_t)t * (S1N * S0K) + tid * 16);
  f[0] = s4[0]; f[1] = s4[1]; f[2] = s4[2]; f[3] = s4[3];
}

__device__ __forceinline__ void store_tile_lds(__bf16* ldsW, int tid,
                                               const float4 f[4]) {
  // native fptrunc -> v_cvt_pk_bf16_f32 (RNE), values land packed in v8bf
  v8bf p0, p1;
  p0[0] = (__bf16)f[0].x; p0[1] = (__bf16)f[0].y;
  p0[2] = (__bf16)f[0].z; p0[3] = (__bf16)f[0].w;
  p0[4] = (__bf16)f[1].x; p0[5] = (__bf16)f[1].y;
  p0[6] = (__bf16)f[1].z; p0[7] = (__bf16)f[1].w;
  p1[0] = (__bf16)f[2].x; p1[1] = (__bf16)f[2].y;
  p1[2] = (__bf16)f[2].z; p1[3] = (__bf16)f[2].w;
  p1[4] = (__bf16)f[3].x; p1[5] = (__bf16)f[3].y;
  p1[6] = (__bf16)f[3].z; p1[7] = (__bf16)f[3].w;
  const int e0  = tid * 16;
  const int row = e0 >> 6;                        // n within tile
  const int col = e0 & 63;                        // k
  v8bf* dst = (v8bf*)&ldsW[row * LDS_ROW + col];  // 16B aligned
  dst[0] = p0;
  dst[1] = p1;
}

__global__ void __launch_bounds__(256)
fastgbn_fused(const float* __restrict__ x,  const float* __restrict__ W1,
              const float* __restrict__ b1, const float* __restrict__ W2,
              const float* __restrict__ b2, float* __restrict__ out,
              int tiles_per_block) {
  // double-buffered W1 tile in bf16 -> one barrier per tile
  __shared__ __bf16 ldsA[S1N * LDS_ROW];
  __shared__ __bf16 ldsB[S1N * LDS_ROW];

  const int tid   = threadIdx.x;
  const int wave  = tid >> 5;        // 8 waves: wave owns batch rows [16w,16w+16)
  const int lane  = tid & 31;
  const int m16   = lane & 15;
  const int khalf = lane >> 4;

  // ---- A fragments: this wave's 16x64 strip of x, bf16, two K-chunks of 32.
  // A 16x32 layout (matches B): lane<16 holds K {0..15}, lane>=16 holds
  // K {16..31} for row M = lane&15.
  v16bf afrag[2];
  {
    const int m = wave * 16 + m16;
#pragma unroll
    for (int kc = 0; kc < 2; ++kc) {
      v16bf tmp;
#pragma unroll
      for (int j = 0; j < 16; ++j) {
        const int k = kc * 32 + khalf * 16 + j;
        tmp[j] = (__bf16)x[m * S0K + k];
      }
      afrag[kc] = tmp;
    }
  }

  const int t0 = blockIdx.x * tiles_per_block;

  // prologue: tile t0 -> registers -> LDS buffer A
  float4 buf[2][4];
  load_tile(W1, t0, tid, buf[0]);
  store_tile_lds(ldsA, tid, buf[0]);
  __syncthreads();

#pragma unroll
  for (int ti = 0; ti < TPB; ++ti) {
    const int t = t0 + ti;
    __bf16* cur = (ti & 1) ? ldsB : ldsA;
    __bf16* nxt = (ti & 1) ? ldsA : ldsB;

    // issue next tile's global loads; they retire during this tile's compute
    if (ti + 1 < TPB) load_tile(W1, t + 1, tid, buf[(ti + 1) & 1]);
    if (t + 2 < T_TILES)
      __builtin_prefetch(W1 + (size_t)(t + 2) * (S1N * S0K) + tid * 16, 0, 1);

    // ---- B fragments + 8 WMMAs ----
    // B 32x16 layout: lane holds 16 consecutive K of column N = nt*16 + m16,
    // K base = kc*32 + khalf*16. 32B read, 16B aligned, conflict-free.
    v16bf bfrag[8];
#pragma unroll
    for (int nt = 0; nt < 4; ++nt) {
      const __bf16* bp = &cur[(nt * 16 + m16) * LDS_ROW + khalf * 16];
#pragma unroll
      for (int kc = 0; kc < 2; ++kc) {
        const v8bf* b8 = (const v8bf*)(bp + kc * 32);
        v8bf lo = b8[0];
        v8bf hi = b8[1];
        bfrag[nt * 2 + kc] = __builtin_shufflevector(lo, hi, 0,1,2,3,4,5,6,7,
                                                     8,9,10,11,12,13,14,15);
      }
    }

    v8f acc[4];
#pragma unroll
    for (int nt = 0; nt < 4; ++nt)
      acc[nt] = (v8f){0.f,0.f,0.f,0.f,0.f,0.f,0.f,0.f};

    // kc outer: adjacent WMMAs target different accumulators (dep distance 4)
#pragma unroll
    for (int kc = 0; kc < 2; ++kc)
#pragma unroll
      for (int nt = 0; nt < 4; ++nt)
        acc[nt] = __builtin_amdgcn_wmma_f32_16x16x32_bf16(
            false, afrag[kc], false, bfrag[nt * 2 + kc],
            (short)0, acc[nt], false, false);

    // scheduler pipeline hint: keep DS reads >= 2 fragments ahead of WMMAs.
    // groups: 8 DS-reads, then {1 WMMA, 2 DS-reads} x4, then 4 WMMAs
    __builtin_amdgcn_sched_group_barrier(0x100, 8, 0);
    __builtin_amdgcn_sched_group_barrier(0x008, 1, 0);
    __builtin_amdgcn_sched_group_barrier(0x100, 2, 0);
    __builtin_amdgcn_sched_group_barrier(0x008, 1, 0);
    __builtin_amdgcn_sched_group_barrier(0x100, 2, 0);
    __builtin_amdgcn_sched_group_barrier(0x008, 1, 0);
    __builtin_amdgcn_sched_group_barrier(0x100, 2, 0);
    __builtin_amdgcn_sched_group_barrier(0x008, 1, 0);
    __builtin_amdgcn_sched_group_barrier(0x100, 2, 0);
    __builtin_amdgcn_sched_group_barrier(0x008, 4, 0);

    // stage tile t+1 into the other LDS buffer; overlaps with the epilogue
    if (ti + 1 < TPB) store_tile_lds(nxt, tid, buf[(ti + 1) & 1]);

    // ---- fused stage 2: bias + ReLU + dot with W2[t,:] ----
    // C layout: acc[nt][r] = h[M = wave*16 + khalf*8 + r][N = nt*16 + m16]
    float ysum[8];
#pragma unroll
    for (int r = 0; r < 8; ++r) ysum[r] = 0.0f;
#pragma unroll
    for (int nt = 0; nt < 4; ++nt) {
      const int c = t * S1N + nt * 16 + m16;
      const float bias = b1[c];
      const float w2v  = W2[c];
#pragma unroll
      for (int r = 0; r < 8; ++r) {
        float h = acc[nt][r] + bias;
        h = fmaxf(h, 0.0f);
        ysum[r] = fmaf(h, w2v, ysum[r]);
      }
    }

    // split-tree reduction over each 16-lane half: 15 ds_swizzle ops total
    float t4[4];
#pragma unroll
    for (int j = 0; j < 4; ++j) {
      float a = ysum[j]     + shflx<1>(ysum[j]);
      float b = ysum[4 + j] + shflx<1>(ysum[4 + j]);
      t4[j] = (lane & 1) ? b : a;
    }
    float t2[2];
#pragma unroll
    for (int j = 0; j < 2; ++j) {
      float a = t4[j]     + shflx<2>(t4[j]);
      float b = t4[2 + j] + shflx<2>(t4[2 + j]);
      t2[j] = (lane & 2) ? b : a;
    }
    {
      float a  = t2[0] + shflx<4>(t2[0]);
      float b  = t2[1] + shflx<4>(t2[1]);
      float t1 = (lane & 4) ? b : a;
      t1 += shflx<8>(t1);
      if ((lane & 8) == 0) {
        // row index from lane bits: r = 4*b0 + 2*b1 + b2
        const int r    = ((lane & 1) << 2) | (lane & 2) | ((lane >> 2) & 1);
        const int brow = wave * 16 + khalf * 8 + r;
        out[(size_t)brow * T_TILES + t] = 2.0f * (t1 + b2[t]);
      }
    }

    if (ti + 1 < TPB) __syncthreads();   // publish nxt buffer for next tile
  }
}

extern "C" void kernel_launch(void* const* d_in, const int* in_sizes, int n_in,
                              void* d_out, int out_size, void* d_ws, size_t ws_size,
                              hipStream_t stream) {
  (void)in_sizes; (void)n_in; (void)out_size; (void)d_ws; (void)ws_size;
  const float* x  = (const float*)d_in[0];
  const float* W1 = (const float*)d_in[1];
  const float* b1 = (const float*)d_in[2];
  const float* W2 = (const float*)d_in[3];
  const float* b2 = (const float*)d_in[4];
  float* out = (float*)d_out;

  dim3 grid(T_TILES / TPB);   // 2048 blocks x 8 tiles
  dim3 block(256);            // 8 waves (wave32)
  fastgbn_fused<<<grid, block, 0, stream>>>(x, W1, b1, W2, b2, out, TPB);
}
